// ModelCRF_66984309948545
// MI455X (gfx1250) — matrix operand order repaired and checked by reading, not compiled, not run on hardware
//
#include <hip/hip_runtime.h>
#include <hip/hip_bf16.h>
#include <stdint.h>

// ---------------------------------------------------------------------------
// CDNA5 (gfx1250) BiLSTM-CRF.
//  - all matmuls on v_wmma_f32_16x16x32_bf16
//  - big GEMMs: 2x2 register blocking + double-buffered LDS tiles filled with
//    GLOBAL_LOAD_ASYNC_TO_LDS_B128 (ASYNCcnt) 
//  - recurrence: h in LDS, c in VGPRs, gate stream prefetched 1 step ahead
// ---------------------------------------------------------------------------

typedef __attribute__((ext_vector_type(16))) __bf16 v16bf;
typedef __attribute__((ext_vector_type(8)))  float  v8f;

union Frag16 { uint4 q[2]; v16bf v; };

#define B_  64
#define T_  512
#define E_  256
#define H_  256
#define BT_ 32768          // B*T
#define G4H_ 1024          // 4*H
#define FC_ 512
#define KTAGS 6
#define KPAD 16
#define NEG_ (-10000.0f)
#define SOS_ 4
#define EOS_ 5

__device__ __forceinline__ unsigned short f32_to_bf16_bits(float f) {
  unsigned int u = __float_as_uint(f);
  u += 0x7FFFu + ((u >> 16) & 1u);          // round-to-nearest-even
  return (unsigned short)(u >> 16);
}
__device__ __forceinline__ float bf16_bits_to_f32(unsigned short h) {
  return __uint_as_float(((unsigned int)h) << 16);
}
__device__ __forceinline__ float sigf(float x) { return 1.0f / (1.0f + __expf(-x)); }

// Async global->LDS copy, 16B per lane (ASYNCcnt-tracked, bypasses VGPRs).
// vdst VGPR holds the wave-relative LDS byte offset (= low 32 bits of the
// generic shared-aperture address).
__device__ __forceinline__ void async_copy_b128(const unsigned short* sh_dst,
                                                const unsigned short* g_src) {
  unsigned lds_off = (unsigned)(uintptr_t)sh_dst;
  asm volatile("global_load_async_to_lds_b128 %0, %1, off"
               :: "v"(lds_off), "v"(g_src) : "memory");
}
__device__ __forceinline__ void wait_async0() {
  asm volatile("s_wait_asynccnt 0" ::: "memory");
}

// A fragment (16x32 bf16).  base -> element [m0][kbase] of row-major A, row
// stride ldk (elements).  Per ISA 7.12.2: lanes 0-15 hold K{0..7,16..23},
// lanes 16-31 hold K{8..15,24..31}, row = lane&15.
__device__ __forceinline__ v16bf load_a_frag(const unsigned short* base, int ldk) {
  int lane = threadIdx.x & 31;
  const unsigned short* p = base + (size_t)(lane & 15) * ldk + ((lane >> 4) << 3);
  Frag16 f;
  f.q[0] = *(const uint4*)(p);
  f.q[1] = *(const uint4*)(p + 16);
  return f.v;
}
// B fragment (32x16).  W stored row-major (N x K); lane = output column n,
// lanes 0-15 hold K{0..7,16..23}, lanes 16-31 hold K{8..15,24..31} (matches A).
__device__ __forceinline__ v16bf load_b_frag(const unsigned short* base, int ldk) {
  int lane = threadIdx.x & 31;
  const unsigned short* p = base + (size_t)(lane & 15) * ldk + ((lane >> 4) << 3);
  Frag16 f;
  f.q[0] = *(const uint4*)(p);
  f.q[1] = *(const uint4*)(p + 16);
  return f.v;
}

// ---------------------------------------------------------------------------
// small prep kernels
// ---------------------------------------------------------------------------
__global__ void cvt_bf16_k(const float* __restrict__ in, unsigned short* __restrict__ out, int n) {
  for (int i = blockIdx.x * blockDim.x + threadIdx.x; i < n; i += gridDim.x * blockDim.x)
    out[i] = f32_to_bf16_bits(in[i]);
}

__global__ void bias_sum_k(const float* a, const float* b, float* out, int n) {
  int i = blockIdx.x * blockDim.x + threadIdx.x;
  if (i < n) out[i] = a[i] + b[i];
}

__global__ void pad_w3_k(const float* __restrict__ w3, const float* __restrict__ b3,
                         unsigned short* __restrict__ w3p, float* __restrict__ b3p) {
  int i = blockIdx.x * blockDim.x + threadIdx.x;
  if (i < KPAD * FC_) {
    int r = i / FC_, c = i % FC_;
    w3p[i] = f32_to_bf16_bits(r < KTAGS ? w3[r * FC_ + c] : 0.0f);
  }
  if (i < KPAD) b3p[i] = (i < KTAGS) ? b3[i] : 0.0f;
}

__global__ void embed_k(const int* __restrict__ text, const float* __restrict__ emb,
                        unsigned short* __restrict__ out) {
  int bt = blockIdx.x, e = threadIdx.x;                 // grid BT_, block E_
  out[(size_t)bt * E_ + e] = f32_to_bf16_bits(emb[(size_t)text[bt] * E_ + e]);
}

// copy with optional reverse-within-length (matches _reverse_within_len)
__global__ void rev_copy_k(const unsigned short* __restrict__ in, unsigned short* __restrict__ out,
                           const int* __restrict__ lengths, int C,
                           int inStride, int outStride, int outColOff, int doReverse) {
  size_t idx = (size_t)blockIdx.x * blockDim.x + threadIdx.x;
  size_t total = (size_t)BT_ * C;
  if (idx >= total) return;
  int c = idx % C;
  int bt = idx / C;
  int t = bt % T_, b = bt / T_;
  int ts = t;
  if (doReverse) { int len = lengths[b]; ts = (t < len) ? (len - 1 - t) : t; }
  out[((size_t)b * T_ + t) * outStride + outColOff + c] =
      in[((size_t)b * T_ + ts) * inStride + c];
}

// ---------------------------------------------------------------------------
// Main WMMA GEMM:  C[M,N] = A[M,K](bf16) * W[N,K](bf16)^T + bias, opt ReLU.
// Block tile 128(M) x 64(N), 8 waves in 4x2, each wave a 32x32 tile (2x2
// accumulators).  A/B tiles double-buffered in LDS, filled with async copies.
// Requires M%128==0, N%64==0, K%32==0.
// ---------------------------------------------------------------------------
__global__ void __launch_bounds__(256)
gemm_bf16_wmma2x2_k(const unsigned short* __restrict__ A,
                    const unsigned short* __restrict__ W,
                    const float* __restrict__ bias,
                    float* __restrict__ outF, unsigned short* __restrict__ outB,
                    int K, int outStride, int bf16Out, int relu) {
  __shared__ unsigned short aT[2][128 * 32];   // 8 KB per buffer
  __shared__ unsigned short bT[2][64 * 32];    // 4 KB per buffer
  int tid  = threadIdx.x;
  int wave = tid >> 5;
  int lane = tid & 31;
  int wm = wave & 3, wn = wave >> 2;           // 4x2 wave grid
  int mBlock = blockIdx.y * 128;
  int nBlock = blockIdx.x * 64;
  const int KC = K >> 5;

  v8f acc[2][2];
#pragma unroll
  for (int i = 0; i < 2; ++i)
#pragma unroll
    for (int j = 0; j < 2; ++j) acc[i][j] = (v8f){0.f,0.f,0.f,0.f,0.f,0.f,0.f,0.f};

  auto stage = [&](int kc, int buf) {
    int k = kc << 5;
    // A tile: 128 rows x 32 cols = 512 x 16B segments
#pragma unroll
    for (int u = 0; u < 2; ++u) {
      int s = tid + 256 * u;
      int row = s >> 2, c8 = (s & 3) << 3;
      async_copy_b128(&aT[buf][row * 32 + c8],
                      A + (size_t)(mBlock + row) * K + k + c8);
    }
    // B tile: 64 rows x 32 cols = 256 x 16B segments
    {
      int row = tid >> 2, c8 = (tid & 3) << 3;
      async_copy_b128(&bT[buf][row * 32 + c8],
                      W + (size_t)(nBlock + row) * K + k + c8);
    }
  };

  stage(0, 0);
  for (int kc = 0; kc < KC; ++kc) {
    int buf = kc & 1;
    wait_async0();
    __syncthreads();                       // tile `buf` visible to all waves
    if (kc + 1 < KC) stage(kc + 1, buf ^ 1);
    v16bf a0 = load_a_frag(&aT[buf][(wm * 32 + 0)  * 32], 32);
    v16bf a1 = load_a_frag(&aT[buf][(wm * 32 + 16) * 32], 32);
    v16bf b0 = load_b_frag(&bT[buf][(wn * 32 + 0)  * 32], 32);
    v16bf b1 = load_b_frag(&bT[buf][(wn * 32 + 16) * 32], 32);
    acc[0][0] = __builtin_amdgcn_wmma_f32_16x16x32_bf16(false, a0, false, b0, (short)0, acc[0][0], false, false);
    acc[0][1] = __builtin_amdgcn_wmma_f32_16x16x32_bf16(false, a0, false, b1, (short)0, acc[0][1], false, false);
    acc[1][0] = __builtin_amdgcn_wmma_f32_16x16x32_bf16(false, a1, false, b0, (short)0, acc[1][0], false, false);
    acc[1][1] = __builtin_amdgcn_wmma_f32_16x16x32_bf16(false, a1, false, b1, (short)0, acc[1][1], false, false);
    // no trailing barrier needed: a wave only re-enters the top barrier after
    // its WMMAs consumed the ds reads, so buffer buf has no pending readers
    // when iteration kc+1 issues its stage into it.
  }

  int hi = lane >> 4;
#pragma unroll
  for (int ti = 0; ti < 2; ++ti)
#pragma unroll
    for (int tj = 0; tj < 2; ++tj) {
      int col = nBlock + wn * 32 + tj * 16 + (lane & 15);
      float bv = bias[col];
#pragma unroll
      for (int v = 0; v < 8; ++v) {
        int row = mBlock + wm * 32 + ti * 16 + v + (hi << 3);
        float x = acc[ti][tj][v] + bv;
        if (relu) x = fmaxf(x, 0.0f);
        size_t off = (size_t)row * outStride + col;
        if (bf16Out) outB[off] = f32_to_bf16_bits(x);
        else         outF[off] = x;
      }
    }
}

// Single-tile GEMM for tiny N (feats head, N=16).
__global__ void __launch_bounds__(256)
gemm_bf16_wmma_k(const unsigned short* __restrict__ A,
                 const unsigned short* __restrict__ W,
                 const float* __restrict__ bias,
                 float* __restrict__ outF, unsigned short* __restrict__ outB,
                 int K, int outStride, int bf16Out, int relu) {
  int wave = threadIdx.x >> 5;
  int lane = threadIdx.x & 31;
  int nt = blockIdx.x * 16;
  int mt = blockIdx.y * 128 + wave * 16;
  const unsigned short* Abase = A + (size_t)mt * K;
  const unsigned short* Wbase = W + (size_t)nt * K;
  v8f acc = {0.f, 0.f, 0.f, 0.f, 0.f, 0.f, 0.f, 0.f};
  for (int k = 0; k < K; k += 32) {
    v16bf a = load_a_frag(Abase + k, K);
    v16bf b = load_b_frag(Wbase + k, K);
    acc = __builtin_amdgcn_wmma_f32_16x16x32_bf16(false, a, false, b,
                                                  (short)0, acc, false, false);
  }
  int col = nt + (lane & 15);
  int hi = lane >> 4;
  float bv = bias[col];
#pragma unroll
  for (int v = 0; v < 8; ++v) {
    int row = mt + v + (hi << 3);
    float x = acc[v] + bv;
    if (relu) x = fmaxf(x, 0.0f);
    size_t off = (size_t)row * outStride + col;
    if (bf16Out) outB[off] = f32_to_bf16_bits(x);
    else         outF[off] = x;
  }
}

// ---------------------------------------------------------------------------
// Sequential LSTM recurrence, one workgroup of 16 waves.
// G holds precomputed  x_t @ W_ih^T + (b_ih+b_hh)  for all (b,t).
// h-state in LDS (bf16), c-state in VGPRs, W_hh streamed from L2.
// Per step each wave runs 4 tile-groups x 4 gates x 8 k-steps = 128 WMMAs.
// ---------------------------------------------------------------------------
__global__ void __launch_bounds__(512)
lstm_dir_wmma_k(const float* __restrict__ G,             // (b*T+t) x 1024
                const unsigned short* __restrict__ Whh,  // 1024 x 256 bf16
                const int* __restrict__ lengths,
                unsigned short* __restrict__ hout,       // ys, masked
                int outStride, int outColOff) {
  __shared__ unsigned short h_sh[B_ * H_];               // 32 KB
  int tid  = threadIdx.x;
  int wave = tid >> 5;
  int lane = tid & 31;
  int col16 = lane & 15;
  int hi = lane >> 4;

  for (int i = tid; i < B_ * H_; i += 512) h_sh[i] = 0;
  float creg[4][8];
#pragma unroll
  for (int r = 0; r < 4; ++r)
#pragma unroll
    for (int v = 0; v < 8; ++v) creg[r][v] = 0.0f;
  __syncthreads();

  for (int t = 0; t < T_; ++t) {
    // prefetch next step's gate rows (64 x 4KB) while computing this step
    if (t + 1 < T_) {
      for (int i = tid; i < 2048; i += 512) {       // 2048 x 128B cachelines
        int b = i >> 5, cl = i & 31;
        __builtin_prefetch(G + ((size_t)b * T_ + t + 1) * G4H_ + cl * 32, 0, 0);
      }
    }
    unsigned short h2reg[4][8];
#pragma unroll
    for (int r = 0; r < 4; ++r) {
      int tg = wave + 16 * r;              // 0..63 unique (m-tile, n-group)
      int mt = (tg & 3) << 4;              // batch-tile base (0,16,32,48)
      int ng = (tg >> 2) << 4;             // hidden-col tile base (0..240)
      v8f acc[4];
#pragma unroll
      for (int g = 0; g < 4; ++g)
#pragma unroll
        for (int v = 0; v < 8; ++v) {
          int b = mt + v + (hi << 3);
          acc[g][v] = G[((size_t)b * T_ + t) * G4H_ + g * H_ + ng + col16];
        }
      for (int k = 0; k < H_; k += 32) {
        v16bf a = load_a_frag(h_sh + mt * H_ + k, H_);
#pragma unroll
        for (int g = 0; g < 4; ++g) {
          v16bf bb = load_b_frag(Whh + (size_t)(g * H_ + ng) * H_ + k, H_);
          acc[g] = __builtin_amdgcn_wmma_f32_16x16x32_bf16(false, a, false, bb,
                                                           (short)0, acc[g], false, false);
        }
      }
      int hcol = ng + col16;
#pragma unroll
      for (int v = 0; v < 8; ++v) {
        int b = mt + v + (hi << 3);
        float iv = sigf(acc[0][v]);
        float fv = sigf(acc[1][v]);
        float gv = tanhf(acc[2][v]);
        float ov = sigf(acc[3][v]);
        float cn = fv * creg[r][v] + iv * gv;
        float hn = ov * tanhf(cn);
        bool m = (t < lengths[b]);
        if (m) creg[r][v] = cn;
        float hprev = bf16_bits_to_f32(h_sh[b * H_ + hcol]);
        h2reg[r][v] = f32_to_bf16_bits(m ? hn : hprev);
        hout[((size_t)b * T_ + t) * outStride + outColOff + hcol] =
            f32_to_bf16_bits(m ? hn : 0.0f);
      }
    }
    __syncthreads();                       // all reads of h_sh done
#pragma unroll
    for (int r = 0; r < 4; ++r) {
      int tg = wave + 16 * r;
      int mt = (tg & 3) << 4;
      int ng = (tg >> 2) << 4;
      int hcol = ng + col16;
#pragma unroll
      for (int v = 0; v < 8; ++v)
        h_sh[(mt + v + (hi << 3)) * H_ + hcol] = h2reg[r][v];
    }
    __syncthreads();
  }
}

// ---------------------------------------------------------------------------
// Viterbi (K=6) : one thread per batch row.
// ---------------------------------------------------------------------------
__global__ void __launch_bounds__(64)
viterbi_k(const float* __restrict__ feats,   // (b*T+t) x KPAD, cols 0..5 valid
          const int* __restrict__ lengths, const float* __restrict__ trans,
          float* __restrict__ score_out, int* __restrict__ path_out,
          int* __restrict__ bptr) {
  int b = threadIdx.x;
  float tr[KTAGS][KTAGS];
  for (int j = 0; j < KTAGS; ++j)
    for (int i = 0; i < KTAGS; ++i) tr[j][i] = trans[j * KTAGS + i];
  float fv[KTAGS];
  for (int j = 0; j < KTAGS; ++j) fv[j] = NEG_;
  fv[SOS_] = 0.0f;
  int len = lengths[b];
  for (int t = 0; t < T_; ++t) {
    bool valid = (t < len);
    float nf[KTAGS]; int bp[KTAGS];
    for (int j = 0; j < KTAGS; ++j) {
      float best = fv[0] + tr[j][0]; int bi = 0;
      for (int i = 1; i < KTAGS; ++i) {
        float s = fv[i] + tr[j][i];
        if (s > best) { best = s; bi = i; }
      }
      nf[j] = best + feats[((size_t)b * T_ + t) * KPAD + j];
      bp[j] = bi;
    }
    for (int j = 0; j < KTAGS; ++j) {
      if (valid) fv[j] = nf[j];
      bptr[((size_t)b * T_ + t) * 8 + j] = valid ? bp[j] : j;
    }
  }
  float best = fv[0] + tr[EOS_][0]; int btag = 0;
  for (int i = 1; i < KTAGS; ++i) {
    float s = fv[i] + tr[EOS_][i];
    if (s > best) { best = s; btag = i; }
  }
  score_out[b] = best;
  int carry = btag;
  for (int t = T_ - 1; t >= 0; --t) {
    int p = carry;
    carry = bptr[((size_t)b * T_ + t) * 8 + p];
    path_out[(size_t)b * T_ + t] = (t < len) ? p : -1;
  }
}

// ---------------------------------------------------------------------------
extern "C" void kernel_launch(void* const* d_in, const int* in_sizes, int n_in,
                              void* d_out, int out_size, void* d_ws, size_t ws_size,
                              hipStream_t stream) {
  (void)in_sizes; (void)n_in; (void)out_size; (void)ws_size;
  const int*   text    = (const int*)d_in[0];
  const int*   lengths = (const int*)d_in[1];
  const float* emb     = (const float*)d_in[2];
  const float *Wih[4], *Whh[4], *bih[4], *bhh[4];
  for (int p = 0; p < 4; ++p) {
    Wih[p] = (const float*)d_in[3 + 4 * p];
    Whh[p] = (const float*)d_in[4 + 4 * p];
    bih[p] = (const float*)d_in[5 + 4 * p];
    bhh[p] = (const float*)d_in[6 + 4 * p];
  }
  const float* w1 = (const float*)d_in[19];
  const float* b1 = (const float*)d_in[20];
  const float* w2 = (const float*)d_in[21];
  const float* b2 = (const float*)d_in[22];
  const float* w3 = (const float*)d_in[23];
  const float* b3 = (const float*)d_in[24];
  const float* trans = (const float*)d_in[25];

  // ---- workspace carve ----
  char* ws = (char*)d_ws;
  size_t off = 0;
  auto alloc = [&](size_t bytes) -> char* {
    char* p = ws + off;
    off += (bytes + 255) & ~(size_t)255;
    return p;
  };
  unsigned short* xemb  = (unsigned short*)alloc((size_t)BT_ * E_ * 2);
  unsigned short* hcat1 = (unsigned short*)alloc((size_t)BT_ * 512 * 2);
  unsigned short* hcat2 = (unsigned short*)alloc((size_t)BT_ * 512 * 2);
  unsigned short* revb  = (unsigned short*)alloc((size_t)BT_ * 512 * 2);
  unsigned short* hdir  = (unsigned short*)alloc((size_t)BT_ * H_ * 2);
  float* G = (float*)alloc((size_t)BT_ * G4H_ * 4);          // 134 MB
  // FC scratch overlaid on G (dead after recurrences)
  unsigned short* fc1   = (unsigned short*)((char*)G);
  unsigned short* fc2   = (unsigned short*)((char*)G + (size_t)BT_ * FC_ * 2);
  float*          feats = (float*)((char*)G + (size_t)2 * BT_ * FC_ * 2);
  unsigned short* wihB[4]; int wihN[4] = {G4H_ * 256, G4H_ * 256, G4H_ * 512, G4H_ * 512};
  for (int p = 0; p < 4; ++p) wihB[p] = (unsigned short*)alloc((size_t)wihN[p] * 2);
  unsigned short* whhB[4];
  for (int p = 0; p < 4; ++p) whhB[p] = (unsigned short*)alloc((size_t)G4H_ * H_ * 2);
  unsigned short* w1B = (unsigned short*)alloc((size_t)FC_ * FC_ * 2);
  unsigned short* w2B = (unsigned short*)alloc((size_t)FC_ * FC_ * 2);
  unsigned short* w3p = (unsigned short*)alloc((size_t)KPAD * FC_ * 2);
  float* b3p   = (float*)alloc(KPAD * 4);
  float* gbias[4];
  for (int p = 0; p < 4; ++p) gbias[p] = (float*)alloc(G4H_ * 4);
  int* bptr = (int*)alloc((size_t)BT_ * 8 * 4);

  // ---- weight prep (bf16 conversion, bias fusion, w3 padding) ----
  for (int p = 0; p < 4; ++p) {
    cvt_bf16_k<<<256, 256, 0, stream>>>(Wih[p], wihB[p], wihN[p]);
    cvt_bf16_k<<<256, 256, 0, stream>>>(Whh[p], whhB[p], G4H_ * H_);
    bias_sum_k<<<4, 256, 0, stream>>>(bih[p], bhh[p], gbias[p], G4H_);
  }
  cvt_bf16_k<<<256, 256, 0, stream>>>(w1, w1B, FC_ * FC_);
  cvt_bf16_k<<<256, 256, 0, stream>>>(w2, w2B, FC_ * FC_);
  pad_w3_k<<<32, 256, 0, stream>>>(w3, b3, w3p, b3p);

  // ---- embedding gather ----
  embed_k<<<BT_, E_, 0, stream>>>(text, emb, xemb);

  const int revBlocks256 = (BT_ * 256 + 255) / 256;
  const int revBlocks512 = (BT_ * 512 + 255) / 256;
  dim3 gGates(G4H_ / 64, BT_ / 128), gFC(FC_ / 64, BT_ / 128), gK(KPAD / 16, BT_ / 128);

  // ---- layer 0 ----
  gemm_bf16_wmma2x2_k<<<gGates, 256, 0, stream>>>(xemb, wihB[0], gbias[0], G, nullptr,
                                                  E_, G4H_, 0, 0);
  lstm_dir_wmma_k<<<1, 512, 0, stream>>>(G, whhB[0], lengths, hcat1, 512, 0);

  rev_copy_k<<<revBlocks256, 256, 0, stream>>>(xemb, revb, lengths, 256, 256, 256, 0, 1);
  gemm_bf16_wmma2x2_k<<<gGates, 256, 0, stream>>>(revb, wihB[1], gbias[1], G, nullptr,
                                                  E_, G4H_, 0, 0);
  lstm_dir_wmma_k<<<1, 512, 0, stream>>>(G, whhB[1], lengths, hdir, 256, 0);
  rev_copy_k<<<revBlocks256, 256, 0, stream>>>(hdir, hcat1, lengths, 256, 256, 512, 256, 1);

  // ---- layer 1 ----
  gemm_bf16_wmma2x2_k<<<gGates, 256, 0, stream>>>(hcat1, wihB[2], gbias[2], G, nullptr,
                                                  512, G4H_, 0, 0);
  lstm_dir_wmma_k<<<1, 512, 0, stream>>>(G, whhB[2], lengths, hcat2, 512, 0);

  rev_copy_k<<<revBlocks512, 256, 0, stream>>>(hcat1, revb, lengths, 512, 512, 512, 0, 1);
  gemm_bf16_wmma2x2_k<<<gGates, 256, 0, stream>>>(revb, wihB[3], gbias[3], G, nullptr,
                                                  512, G4H_, 0, 0);
  lstm_dir_wmma_k<<<1, 512, 0, stream>>>(G, whhB[3], lengths, hdir, 256, 0);
  rev_copy_k<<<revBlocks256, 256, 0, stream>>>(hdir, hcat2, lengths, 256, 256, 512, 256, 1);

  // ---- FC head (note: fc1/fc2/feats live in the now-dead G region) ----
  gemm_bf16_wmma2x2_k<<<gFC, 256, 0, stream>>>(hcat2, w1B, b1, nullptr, fc1, 512, FC_, 1, 1);
  gemm_bf16_wmma2x2_k<<<gFC, 256, 0, stream>>>(fc1, w2B, b2, nullptr, fc2, FC_, FC_, 1, 1);
  gemm_bf16_wmma_k<<<gK, 256, 0, stream>>>(fc2, w3p, b3p, feats, nullptr, FC_, KPAD, 0, 0);

  // ---- CRF Viterbi ----
  viterbi_k<<<1, 64, 0, stream>>>(feats, lengths, trans,
                                  (float*)d_out, ((int*)d_out) + B_, bptr);
}
